// BauAttn_61237643707095
// MI455X (gfx1250) — matrix operand order repaired and checked by reading, not compiled, not run on hardware
//
#include <hip/hip_runtime.h>
#include <math.h>

// ---------- problem constants ----------
#define LSEQ 2048
#define BSZ  64
#define DENC 512
#define ASZ  64
#define KPAD 520            // 512 + 8 pad (keeps 16B alignment, staggers LDS banks)

// ---------- workspace layout (bytes) ----------
#define OFF_WT     0                               // bf16 [2][64][520] hi/lo of W_enc^T
#define SZ_WT      (2 * 64 * KPAD * 2)             // 133120
#define OFF_HID    (OFF_WT + SZ_WT)                // f32 [64][64] hid+b_attn (contiguous after WT!)
#define SZ_HID     (64 * 64 * 4)                   // 16384
#define OFF_STATS  (OFF_HID + SZ_HID)              // f32 [64][2] {max, 1/sum}
#define SZ_STATS   (64 * 2 * 4)
#define OFF_SCORES (OFF_STATS + SZ_STATS + 96)     // f32 [64][2048], keep 16B align
#define SZ_SCORES  (BSZ * LSEQ * 4)                // 524288
#define OFF_PART   (OFF_SCORES + SZ_SCORES)        // f32 [64][16][512]
#define SZ_PART    (BSZ * 16 * DENC * 4)           // 2097152

#define SMEM_BYTES (SZ_WT + SZ_HID)                // 149504 < 320KB WGP LDS
static_assert(OFF_HID == OFF_WT + SZ_WT, "W and hid must be contiguous for single TDM copy");
static_assert(SMEM_BYTES % 8 == 0, "TDM copy uses 8-byte elements");

typedef __bf16 bf16_t;
typedef __bf16 v16bf __attribute__((ext_vector_type(16)));
typedef __bf16 v8bf  __attribute__((ext_vector_type(8)));
typedef float  v8f   __attribute__((ext_vector_type(8)));
typedef float  v4f   __attribute__((ext_vector_type(4)));
typedef unsigned int v4u __attribute__((ext_vector_type(4)));
typedef int    v4i   __attribute__((ext_vector_type(4)));
typedef int    v8i   __attribute__((ext_vector_type(8)));

// ---------------- kernel 1: hid[b][n] = b_attn[n] + hidden[b,:] @ W_hidden[:,n]
__global__ void k_hid(const float* __restrict__ hidden, const float* __restrict__ Wh,
                      const float* __restrict__ battn, float* __restrict__ hid_pb) {
    int b = blockIdx.x, n = threadIdx.x;
    float acc = battn[n];
    for (int d = 0; d < DENC; ++d)
        acc = fmaf(hidden[b * DENC + d], Wh[d * ASZ + n], acc);
    hid_pb[b * ASZ + n] = acc;
}

// ---------------- kernel 2: split W_enc (K x N) into bf16 hi/lo stored [N][KPAD]
__global__ void k_pack(const float* __restrict__ We, bf16_t* __restrict__ wt) {
    int i = blockIdx.x * blockDim.x + threadIdx.x;   // 0 .. 64*512-1
    int n = i >> 9, k = i & 511;
    float f = We[k * ASZ + n];
    bf16_t hi = (bf16_t)f;
    bf16_t lo = (bf16_t)(f - (float)hi);
    wt[n * KPAD + k]              = hi;
    wt[64 * KPAD + n * KPAD + k]  = lo;
}

// ---------------- kernel 3: scores[b][l] = v . tanh(enc@W_enc + hid)  via WMMA bf16 hi/lo
__global__ void __launch_bounds__(256)
k_scores(const float* __restrict__ enc, const bf16_t* __restrict__ wt,
         const float* __restrict__ hid_pb, const float* __restrict__ vvec,
         float* __restrict__ scores) {
    extern __shared__ char smem[];
    bf16_t* sW = (bf16_t*)smem;                 // [2][64][KPAD]
    float*  sH = (float*)(smem + SZ_WT);        // [64][64]

    // ---- stage W(hi/lo)+hid into LDS ----
#if __has_builtin(__builtin_amdgcn_tensor_load_to_lds)
    // Tensor Data Mover: one 1-D D# moves the whole 149.5 KB block image
    // (W hi/lo + hid are contiguous in ws and in LDS). TENSORcnt-tracked,
    // no VGPR round-trip. Only wave 0 issues; others released by barrier.
    if (threadIdx.x < 32) {
        const unsigned lds_off = (unsigned)(uintptr_t)smem;      // flat-shared low 32b = LDS offset
        const unsigned long long ga = (unsigned long long)(uintptr_t)wt;
        const unsigned nelem = SMEM_BYTES / 8;                   // 8-byte elements (18688)

        v4u g0;
        g0[0] = 1u;                                              // count=1, user flags 0
        g0[1] = lds_off;                                         // lds_addr [63:32]
        g0[2] = (unsigned)(ga & 0xffffffffu);                    // global_addr [95:64]
        g0[3] = (unsigned)((ga >> 32) & 0x01ffffffu) | (2u << 30); // addr[56:32] | type=2

        v8i g1;
        g1[0] = (int)(3u << 16);                 // workgroup_mask=0 | data_size=3 (8B)
        g1[1] = (int)((nelem & 0xffffu) << 16);  // tensor_dim0[15:0] at bits[63:48]
        g1[2] = (int)((nelem >> 16) | (1u << 16)); // tensor_dim0[31:16] | tensor_dim1=1
        g1[3] = (int)((nelem & 0xffffu) << 16);  // tensor_dim1 hi=0 | tile_dim0 at [127:112]
        g1[4] = (int)((nelem >> 16) == 0 ? 1 : 1); // tile_dim1=1 (tile_dim2=0)
        g1[5] = (int)nelem;                      // tensor_dim0_stride[31:0]
        g1[6] = 0;                               // stride0 hi | stride1 lo
        g1[7] = 0;                               // stride1 hi

        v4i z4 = {0, 0, 0, 0};
#if __clang_major__ >= 23
        v8i z8 = {0, 0, 0, 0, 0, 0, 0, 0};
        __builtin_amdgcn_tensor_load_to_lds(g0, g1, z4, z4, z8, 0);
#else
        __builtin_amdgcn_tensor_load_to_lds(g0, g1, z4, z4, 0);
#endif
        __builtin_amdgcn_s_wait_tensorcnt(0);
    }
#else
    // Fallback: cooperative copy through VGPRs
    {
        const uint4* gW = (const uint4*)wt;
        uint4* dW = (uint4*)sW;
        for (int i = threadIdx.x; i < SZ_WT / 16; i += 256) dW[i] = gW[i];
        const uint4* gH = (const uint4*)hid_pb;
        uint4* dH = (uint4*)sH;
        for (int i = threadIdx.x; i < SZ_HID / 16; i += 256) dH[i] = gH[i];
    }
#endif
    __syncthreads();

    const int lane = threadIdx.x & 31;
    const int wave = threadIdx.x >> 5;          // 8 waves
    const int r    = lane & 15;                 // row-in-tile / col-in-tile
    const int kh   = lane >> 4;                 // half-wave selector
    const int l    = blockIdx.x * 8 + wave;     // one sequence row per wave

    float vreg[4];
#pragma unroll
    for (int nt = 0; nt < 4; ++nt) vreg[nt] = vvec[nt * 16 + r];

    for (int q = 0; q < 4; ++q) {               // 4 M-tiles of 16 (b0 = 16q)
        const int b0 = q * 16;
        // A row base: M = l*64 + b0 + r ; 16-bit A layout: lanes<16 -> K {0..7,16..23}
        const float* aptr = enc + ((size_t)(l * 64 + b0 + r)) * DENC + kh * 8;

        v8f acc[4] = {};
        for (int kc = 0; kc < 16; ++kc) {       // K chunks of 32
            const int kb = kc * 32;
            // enc is a 256MB read-once stream: non-temporal, keep W/scores in L2
            v4f f0 = __builtin_nontemporal_load((const v4f*)(aptr + kb));
            v4f f1 = __builtin_nontemporal_load((const v4f*)(aptr + kb + 4));
            v4f f2 = __builtin_nontemporal_load((const v4f*)(aptr + kb + 16));
            v4f f3 = __builtin_nontemporal_load((const v4f*)(aptr + kb + 20));

            v16bf ahi, alo;
#pragma unroll
            for (int e = 0; e < 16; ++e) {
                float f = (e < 4) ? f0[e] : (e < 8) ? f1[e - 4]
                        : (e < 12) ? f2[e - 8] : f3[e - 12];
                bf16_t h = (bf16_t)f;
                ahi[e] = h;
                alo[e] = (bf16_t)(f - (float)h);
            }

            const int kofs = kb + kh * 8;       // B layout: lanes<16 K={0..7,16..23}, lanes>=16 K={8..15,24..31}
#pragma unroll
            for (int nt = 0; nt < 4; ++nt) {
                const bf16_t* bhp = sW + (size_t)(nt * 16 + r) * KPAD + kofs;
                const bf16_t* blp = bhp + 64 * KPAD;
                v8bf bh0 = *(const v8bf*)bhp;
                v8bf bh1 = *(const v8bf*)(bhp + 16);
                v8bf bl0 = *(const v8bf*)blp;
                v8bf bl1 = *(const v8bf*)(blp + 16);
                v16bf bhi, blo;
#pragma unroll
                for (int e = 0; e < 8; ++e) {
                    bhi[e] = bh0[e]; bhi[e + 8] = bh1[e];
                    blo[e] = bl0[e]; blo[e + 8] = bl1[e];
                }
                acc[nt] = __builtin_amdgcn_wmma_f32_16x16x32_bf16(
                    false, ahi, false, bhi, (short)0, acc[nt], false, false);
                acc[nt] = __builtin_amdgcn_wmma_f32_16x16x32_bf16(
                    false, ahi, false, blo, (short)0, acc[nt], false, false);
                acc[nt] = __builtin_amdgcn_wmma_f32_16x16x32_bf16(
                    false, alo, false, bhi, (short)0, acc[nt], false, false);
            }
        }

        // epilogue: + hid, tanh, dot v over N, reduce across 16 lanes
        float s[8];
#pragma unroll
        for (int j = 0; j < 8; ++j) s[j] = 0.f;
#pragma unroll
        for (int nt = 0; nt < 4; ++nt) {
#pragma unroll
            for (int j = 0; j < 8; ++j) {
                const int brow = b0 + j + 8 * kh;            // C layout: VGPR j -> M=j / M=8+j
                float x = acc[nt][j] + sH[brow * ASZ + nt * 16 + r];
                s[j] = fmaf(tanhf(x), vreg[nt], s[j]);
            }
        }
#pragma unroll
        for (int j = 0; j < 8; ++j) {
            float t = s[j];
            t += __shfl_xor(t, 1, 32);
            t += __shfl_xor(t, 2, 32);
            t += __shfl_xor(t, 4, 32);
            t += __shfl_xor(t, 8, 32);
            if (r == 0)                                       // lanes 0 and 16
                scores[(size_t)(b0 + 8 * kh + j) * LSEQ + l] = t;
        }
    }
}

// ---------------- kernel 4: per-b masked softmax stats {max, 1/sum}
__global__ void k_stats(const float* __restrict__ scores, const int* __restrict__ lens,
                        float* __restrict__ stats) {
    __shared__ float red[256];
    const int b = blockIdx.x;
    const int len = lens[b];
    const float* sc = scores + (size_t)b * LSEQ;
    float m = -INFINITY;
    for (int l = threadIdx.x; l < len; l += 256) m = fmaxf(m, sc[l]);
    red[threadIdx.x] = m; __syncthreads();
    for (int s = 128; s > 0; s >>= 1) {
        if (threadIdx.x < s) red[threadIdx.x] = fmaxf(red[threadIdx.x], red[threadIdx.x + s]);
        __syncthreads();
    }
    m = red[0]; __syncthreads();
    float sum = 0.f;
    for (int l = threadIdx.x; l < len; l += 256) sum += expf(sc[l] - m);
    red[threadIdx.x] = sum; __syncthreads();
    for (int s = 128; s > 0; s >>= 1) {
        if (threadIdx.x < s) red[threadIdx.x] += red[threadIdx.x + s];
        __syncthreads();
    }
    if (threadIdx.x == 0) { stats[b * 2] = m; stats[b * 2 + 1] = 1.0f / red[0]; }
}

// ---------------- kernel 5: partial[b][chunk][d] = sum_{l in chunk, l<len} w[l]*enc[l,b,d]
__global__ void k_wsum(const float* __restrict__ enc, const float* __restrict__ scores,
                       const float* __restrict__ stats, const int* __restrict__ lens,
                       float* __restrict__ partial) {
    __shared__ float wl[128];
    const int b = blockIdx.y, chunk = blockIdx.x;
    const int lbase = chunk * 128;
    const int len = lens[b];
    const float m = stats[b * 2], rinv = stats[b * 2 + 1];
    int cnt = len - lbase;
    if (cnt > 128) cnt = 128;
    if (cnt < 0)   cnt = 0;
    const int t = threadIdx.x;
    if (t < cnt) wl[t] = expf(scores[(size_t)b * LSEQ + lbase + t] - m) * rinv;
    __syncthreads();
    v4f acc = {0.f, 0.f, 0.f, 0.f};
    for (int i = 0; i < cnt; ++i) {
        const float w = wl[i];
        const v4f ev = __builtin_nontemporal_load(
            (const v4f*)(enc + ((size_t)(lbase + i) * BSZ + b) * DENC + t * 4));
        acc += ev * w;
    }
    *(v4f*)(partial + ((size_t)(b * 16 + chunk)) * DENC + t * 4) = acc;
}

// ---------------- kernel 6: out[b][d] = sum_chunk partial
__global__ void k_final(const float* __restrict__ partial, float* __restrict__ out) {
    const int b = blockIdx.x, t = threadIdx.x;
    v4f acc = {0.f, 0.f, 0.f, 0.f};
    for (int c = 0; c < 16; ++c)
        acc += *(const v4f*)(partial + ((size_t)(b * 16 + c)) * DENC + t * 4);
    *(v4f*)(out + (size_t)b * DENC + t * 4) = acc;
}

extern "C" void kernel_launch(void* const* d_in, const int* in_sizes, int n_in,
                              void* d_out, int out_size, void* d_ws, size_t ws_size,
                              hipStream_t stream) {
    const float* enc      = (const float*)d_in[0];
    const int*   lens     = (const int*)  d_in[1];
    const float* hidden   = (const float*)d_in[2];
    const float* W_enc    = (const float*)d_in[3];
    const float* b_attn   = (const float*)d_in[4];
    const float* W_hidden = (const float*)d_in[5];
    const float* vvec     = (const float*)d_in[6];
    float* out = (float*)d_out;

    char* ws = (char*)d_ws;
    bf16_t* wt      = (bf16_t*)(ws + OFF_WT);
    float*  hid_pb  = (float*) (ws + OFF_HID);
    float*  stats   = (float*) (ws + OFF_STATS);
    float*  scores  = (float*) (ws + OFF_SCORES);
    float*  partial = (float*) (ws + OFF_PART);

    (void)in_sizes; (void)n_in; (void)out_size; (void)ws_size;

    hipFuncSetAttribute((const void*)k_scores,
                        hipFuncAttributeMaxDynamicSharedMemorySize, SMEM_BYTES);

    k_hid   <<<64, 64, 0, stream>>>(hidden, W_hidden, b_attn, hid_pb);
    k_pack  <<<128, 256, 0, stream>>>(W_enc, wt);
    k_scores<<<256, 256, SMEM_BYTES, stream>>>(enc, wt, hid_pb, vvec, scores);
    k_stats <<<64, 256, 0, stream>>>(scores, lens, stats);
    k_wsum  <<<dim3(16, 64), 128, 0, stream>>>(enc, scores, stats, lens, partial);
    k_final <<<64, 128, 0, stream>>>(partial, out);
}
